// LinearAttention2_81561428951277
// MI455X (gfx1250) — matrix operand order repaired and checked by reading, not compiled, not run on hardware
//
#include <hip/hip_runtime.h>
#include <cstdint>
#include <cstddef>

// ---------------- feature detection ----------------
#ifndef __has_builtin
#define __has_builtin(x) 0
#endif

#if __has_builtin(__builtin_amdgcn_tensor_load_to_lds) && __has_builtin(__builtin_amdgcn_s_wait_tensorcnt)
#define HAVE_TDM 1
#endif
#if __has_builtin(__builtin_amdgcn_global_load_async_to_lds_b128)
#define HAVE_ASYNC 1
#endif

#if defined(__has_include)
#if __has_include(<hip/amd_detail/amd_gfx1250_TDM.h>)
#define TDM_6ARG 1   // therock-10.0 headers => 6-arg tensor_load_to_lds builtin
#endif
#endif

// ---------------- constants ----------------
constexpr int  Bk   = 8;
constexpr int  Hk   = 120;
constexpr int  Wk   = 160;
constexpr int  Ck   = 96;
constexpr int  Nk   = Hk * Wk;           // 19200
constexpr int  BNk  = Bk * Nk;           // 153600
constexpr int  NHEAD = 3;
constexpr int  Dk   = 32;

typedef __bf16 v16bf __attribute__((ext_vector_type(16)));
typedef float  v8f   __attribute__((ext_vector_type(8)));
typedef unsigned int u32x4 __attribute__((ext_vector_type(4)));
typedef int          i32x4 __attribute__((ext_vector_type(4)));
typedef int          i32x8 __attribute__((ext_vector_type(8)));

union ABf { v16bf v; uint32_t u[8]; };
union CFr { v8f v; float f[8]; };

static __device__ __forceinline__ uint16_t f2bf(float f) {
    uint32_t u = __float_as_uint(f);
    uint32_t r = u + 0x7FFFu + ((u >> 16) & 1u);   // round-to-nearest-even
    return (uint16_t)(r >> 16);
}
static __device__ __forceinline__ float bf2f(uint16_t h) {
    return __uint_as_float(((uint32_t)h) << 16);
}
static __device__ __forceinline__ uint32_t pack2(uint16_t lo, uint16_t hi) {
    return (uint32_t)lo | ((uint32_t)hi << 16);
}
// generic LDS pointer -> LDS byte offset (low 32 bits of the LDS aperture addr)
static __device__ __forceinline__ uint32_t lds_off_of(const void* p) {
    return (uint32_t)(uintptr_t)p;
}

#if defined(HAVE_TDM)
// Issue a 2-D TDM tile load: tile_d1 rows x tile_d0 elements, 2-byte elements,
// row stride `stride_elems`, packed contiguously into LDS at lds_off.
static __device__ __forceinline__ void tdm_load_tile_2d_b16(
    uint32_t lds_off, const void* gaddr,
    uint32_t tile_d0, uint32_t tile_d1, uint32_t stride_elems)
{
    const uint64_t ga = (uint64_t)(uintptr_t)gaddr;
    u32x4 g0;
    g0.x = 1u;                                   // count=1 valid user descriptor
    g0.y = lds_off;                              // lds_addr
    g0.z = (uint32_t)ga;                         // global_addr[31:0]
    g0.w = (uint32_t)((ga >> 32) & 0x1FFFFFFu) | (2u << 30);   // addr[56:32], type=2
    i32x8 g1;
    const uint32_t td0 = tile_d0, td1 = tile_d1;
    g1[0] = (int)(1u << 16);                                         // data_size=1 (2B)
    g1[1] = (int)((td0 & 0xFFFFu) << 16);                            // tensor_dim0 lo
    g1[2] = (int)(((td0 >> 16) & 0xFFFFu) | ((td1 & 0xFFFFu) << 16));// td0 hi | td1 lo
    g1[3] = (int)(((td1 >> 16) & 0xFFFFu) | ((tile_d0 & 0xFFFFu) << 16)); // tile_dim0
    g1[4] = (int)(tile_d1 & 0xFFFFu);                                // tile_dim1, tile_dim2=0
    g1[5] = (int)stride_elems;                                       // dim0_stride[31:0]
    g1[6] = 0;                                                       // stride hi / dim1_stride
    g1[7] = 0;
    i32x4 z4 = {0, 0, 0, 0};
#if defined(TDM_6ARG)
    i32x8 z8 = {0, 0, 0, 0, 0, 0, 0, 0};
    __builtin_amdgcn_tensor_load_to_lds(g0, g1, z4, z4, z8, 0);
#else
    __builtin_amdgcn_tensor_load_to_lds(g0, g1, z4, z4, 0);
#endif
}
#endif

#if defined(HAVE_ASYNC)
// Diagnostic from probe: param 0 is `int __attribute__((vector_size(16))) addrspace(1)*`
typedef int v4i_t __attribute__((vector_size(16)));
typedef __attribute__((address_space(1))) v4i_t gv4i;
typedef __attribute__((address_space(3))) v4i_t lv4i;
static __device__ __forceinline__ void async_copy_b128(const void* gsrc, uint32_t lds_off) {
    __builtin_amdgcn_global_load_async_to_lds_b128(
        (gv4i*)(uintptr_t)gsrc, (lv4i*)(uintptr_t)lds_off, 0, 0);
}
static __device__ __forceinline__ void wait_async0() {
#if __has_builtin(__builtin_amdgcn_s_wait_asynccnt)
    __builtin_amdgcn_s_wait_asynccnt(0);
#else
    asm volatile("s_wait_asynccnt 0" ::: "memory");
#endif
}
#endif

// =====================================================================
// Kernel 0: prep.  Zero ksum[2]; build W fragments in bf16, pre-swizzled
// into WMMA B-fragment order: Wfrag[((ks*18+ot)*32+lane)*8+i] so that each
// lane of the projection kernel loads 8 contiguous dwords per fragment.
// =====================================================================
__global__ void prep_kernel(const float* __restrict__ Wqk,
                            const float* __restrict__ Wv,
                            uint32_t* __restrict__ Wfrag,
                            float* __restrict__ ksum0,
                            float* __restrict__ ksum1)
{
    int t = threadIdx.x;
    for (int idx = t; idx < 3 * 18 * 32 * 8; idx += 256) {
        int i    = idx & 7;
        int r    = idx >> 3;
        int lane = r & 31;
        int q    = r >> 5;
        int ot   = q % 18;
        int ks   = q / 18;
        int m  = lane & 15;
        int hi = lane >> 4;
        int o  = ot * 16 + m;               // output column (0..287)
        int kk = ks * 32 + hi * 16 + 2 * i; // K index (0..95), even
        float w0 = (o < 192) ? Wqk[o * 96 + kk]     : Wv[(o - 192) * 96 + kk];
        float w1 = (o < 192) ? Wqk[o * 96 + kk + 1] : Wv[(o - 192) * 96 + kk + 1];
        Wfrag[idx] = pack2(f2bf(w0), f2bf(w1));
    }
    for (int idx = t; idx < Bk * Ck; idx += 256) { ksum0[idx] = 0.f; ksum1[idx] = 0.f; }
}

// =====================================================================
// Kernel 1: fused projection. One block = 128 rows of (b*n) x all 288 cols.
//   Y = X @ [Wqk;Wv]^T + bias ; q,k -> elu()+1 ; RoPE(q,k) ; ksum += k
// Stores qf (bf16, non-roped, for z), qr/kr (bf16, roped), v (f32).
// =====================================================================
__global__ __launch_bounds__(256) void proj_kernel(
    const float*   __restrict__ X,
    const uint32_t* __restrict__ Wfrag,
    const float*   __restrict__ bqk,
    const float*   __restrict__ bv,
    uint16_t* __restrict__ qf,
    uint16_t* __restrict__ qr,
    uint16_t* __restrict__ kr,
    float*    __restrict__ Vout,
    float*    __restrict__ ksum)
{
    __shared__ __align__(16) uint16_t sX[128 * 96];   // 24.5 KB bf16 X tile
    __shared__ float sHc[2 * 24], sHs[2 * 24];        // h-axis rope (<=2 y rows/tile)
    __shared__ float sWc[128 * 24], sWs[128 * 24];    // w-axis rope per local row

    const int t    = threadIdx.x;
    const int row0 = blockIdx.x * 128;                // global row in [0,BN)
    const int bb   = row0 / Nk;
    const int nn0  = row0 % Nk;
    const int x0   = nn0 % Wk;
    const int y0   = nn0 / Wk;

    // stage X tile -> bf16 LDS (coalesced float4)
    {
        const float4* src = (const float4*)(X + (size_t)row0 * Ck);
        for (int i = t; i < 128 * 96 / 4; i += 256) {
            float4 v = src[i];
            sX[i * 4 + 0] = f2bf(v.x); sX[i * 4 + 1] = f2bf(v.y);
            sX[i * 4 + 2] = f2bf(v.z); sX[i * 4 + 3] = f2bf(v.w);
        }
    }
    // rope tables.  theta_j = 10000^(-j/24) = exp(-j*ln(1e4)/24)
    for (int i = t; i < 2 * 24; i += 256) {
        int yy = y0 + i / 24, j = i % 24;
        float theta = __expf(-(float)j * (9.210340371976184f / 24.0f));
        float s, c; __sincosf((float)yy * theta, &s, &c);
        sHc[i] = c; sHs[i] = s;
    }
    for (int i = t; i < 128 * 24; i += 256) {
        int rl = i / 24, j = i % 24;
        int xx = (x0 + rl) % Wk;
        float theta = __expf(-(float)j * (9.210340371976184f / 24.0f));
        float s, c; __sincosf((float)xx * theta, &s, &c);
        sWc[i] = c; sWs[i] = s;
    }
    __syncthreads();

    const int wv   = t >> 5;
    const int lane = t & 31;
    const int m    = lane & 15;
    const int hi   = lane >> 4;
    const int r0   = wv * 16;                // wave's 16 rows inside tile
    const int kbaseA = hi * 8;

    // A fragments for K-steps 0,32,64 (16x32 bf16, k = 16*hi + s, same order as Wfrag)
    ABf afrag[3];
#pragma unroll
    for (int ks = 0; ks < 3; ++ks) {
        const uint32_t* xr = (const uint32_t*)&sX[(r0 + m) * 96];
#pragma unroll
        for (int i = 0; i < 8; ++i) {
            int kk = ks * 32 + hi * 16 + 2 * i;
            afrag[ks].u[i] = xr[kk >> 1];
        }
    }

    for (int ot = 0; ot < 18; ++ot) {
        const int o = ot * 16 + m;           // this lane's output column
        CFr c;
#pragma unroll
        for (int i = 0; i < 8; ++i) c.f[i] = 0.f;
#pragma unroll
        for (int ks = 0; ks < 3; ++ks) {
            ABf bfrag;
            const uint32_t* wf = Wfrag + ((size_t)(ks * 18 + ot) * 32 + lane) * 8;
#pragma unroll
            for (int i = 0; i < 8; ++i) bfrag.u[i] = wf[i];
            c.v = __builtin_amdgcn_wmma_f32_16x16x32_bf16(
                false, afrag[ks].v, false, bfrag.v, (short)0, c.v, false, false);
        }

        const float bias = (o < 192) ? bqk[o] : bv[o - 192];
        if (ot >= 12) {
            // V columns: just bias, store f32
#pragma unroll
            for (int r = 0; r < 8; ++r) {
                int rl = r0 + r + hi * 8;
                Vout[(size_t)(row0 + rl) * Ck + (o - 192)] = c.f[r] + bias;
            }
        } else {
            // q/k columns: elu()+1
            float act[8]; float ssum = 0.f;
#pragma unroll
            for (int r = 0; r < 8; ++r) {
                float v = c.f[r] + bias;
                act[r] = (v > 0.f) ? (v + 1.f) : __expf(v);
                ssum += act[r];
            }
            const bool isQ = (ot < 6);
            const int  ch  = isQ ? o : (o - 96);
            if (isQ) {
#pragma unroll
                for (int r = 0; r < 8; ++r) {
                    int rl = r0 + r + hi * 8;
                    qf[(size_t)(row0 + rl) * Ck + ch] = f2bf(act[r]);
                }
            } else {
                float tot = ssum + __shfl_xor(ssum, 16, 32);
                if (hi == 0) atomicAdd(&ksum[bb * Ck + ch], tot);
            }
            // RoPE: pair partner lives in adjacent lane (adjacent column)
            const int  j    = ch >> 1;
            const bool even = ((ch & 1) == 0);
            uint16_t outv[8];
#pragma unroll
            for (int r = 0; r < 8; ++r) {
                float other = __shfl_xor(act[r], 1, 32);
                int rl = r0 + r + hi * 8;
                float cs, sn;
                if (j < 24) {
                    int yloc = ((x0 + rl) >= Wk) ? 1 : 0;
                    cs = sHc[yloc * 24 + j]; sn = sHs[yloc * 24 + j];
                } else {
                    cs = sWc[rl * 24 + (j - 24)]; sn = sWs[rl * 24 + (j - 24)];
                }
                float ov = even ? (cs * act[r] - sn * other)
                                : (sn * other + cs * act[r]);
                outv[r] = f2bf(ov);
            }
            uint16_t* dst = isQ ? qr : kr;
#pragma unroll
            for (int r = 0; r < 8; ++r) {
                int rl = r0 + r + hi * 8;
                dst[(size_t)(row0 + rl) * Ck + ch] = outv[r];
            }
        }
    }
}

// =====================================================================
// Kernel 2: kv[b,h] = (1/n) * kr^T @ v   (32x32 per (b,head))
// One block per (b,head); 8 waves each reduce 75 chunks of 32 rows.
// k-tile staged via TDM (preferred) / async-to-LDS / manual fallback.
// =====================================================================
__global__ __launch_bounds__(256) void kv_kernel(
    const uint16_t* __restrict__ krp,
    const float*    __restrict__ Vin,
    float*          __restrict__ kvout)
{
    __shared__ __align__(16) uint16_t sK[8][32 * 32];
    __shared__ __align__(16) uint16_t sV[8][32 * 32];
    __shared__ float sAcc[32 * 32];

    const int t = threadIdx.x, wv = t >> 5, lane = t & 31;
    const int bb = blockIdx.x / NHEAD, head = blockIdx.x % NHEAD;
    for (int i = t; i < 1024; i += 256) sAcc[i] = 0.f;
    __syncthreads();

    CFr c00, c01, c10, c11;
#pragma unroll
    for (int i = 0; i < 8; ++i) { c00.f[i] = 0; c01.f[i] = 0; c10.f[i] = 0; c11.f[i] = 0; }

    const size_t base = (size_t)bb * Nk;
    const int m = lane & 15, hi = lane >> 4;
    const int kbaseA = hi * 8, kbaseB = hi * 16;

    for (int ch = wv; ch < Nk / 32; ch += 8) {
        const int n0 = ch * 32;
        {   // prefetch next chunk for this wave
            int nnx = (ch + 8) * 32;
            if (nnx < Nk) {
                __builtin_prefetch(&krp[(base + nnx + lane) * Ck + head * Dk], 0, 1);
                __builtin_prefetch(&Vin[(base + nnx + lane) * Ck + head * Dk], 0, 1);
            }
        }
        // ---- stage k tile (32 rows x 32 bf16, row stride 96 elem) ----
#if defined(HAVE_TDM)
        tdm_load_tile_2d_b16(lds_off_of(&sK[wv][0]),
                             &krp[(base + n0) * Ck + head * Dk],
                             /*tile_d0=*/32, /*tile_d1=*/32, /*stride=*/Ck);
#elif defined(HAVE_ASYNC)
        {
            const char* gsrc = (const char*)&krp[(base + n0 + lane) * Ck + head * Dk];
            uint32_t dlo = lds_off_of(&sK[wv][lane * 32]);
#pragma unroll
            for (int q = 0; q < 4; ++q) async_copy_b128(gsrc + q * 16, dlo + q * 16);
        }
#else
        {
            const uint4* ks = (const uint4*)&krp[(base + n0 + lane) * Ck + head * Dk];
            uint4* kd = (uint4*)&sK[wv][lane * 32];
            kd[0] = ks[0]; kd[1] = ks[1]; kd[2] = ks[2]; kd[3] = ks[3];
        }
#endif
        // ---- stage v tile (f32 -> bf16 conversion, manual) ----
        {
            const float4* vs = (const float4*)&Vin[(base + n0 + lane) * Ck + head * Dk];
            uint16_t* vd = &sV[wv][lane * 32];
#pragma unroll
            for (int q = 0; q < 8; ++q) {
                float4 f = vs[q];
                vd[q * 4 + 0] = f2bf(f.x); vd[q * 4 + 1] = f2bf(f.y);
                vd[q * 4 + 2] = f2bf(f.z); vd[q * 4 + 3] = f2bf(f.w);
            }
        }
        __builtin_amdgcn_wave_barrier();
#if defined(HAVE_TDM)
        __builtin_amdgcn_s_wait_tensorcnt(0);
#elif defined(HAVE_ASYNC)
        wait_async0();
#endif
        asm volatile("s_wait_dscnt 0" ::: "memory");

        // A = k^T (rows=d, K=n), B = v (K=n, cols=e)
        ABf a0, a1, b0, b1;
#pragma unroll
        for (int i = 0; i < 8; ++i) {
            int kk = kbaseB + 2 * i;                           // n index
            a0.u[i] = pack2(sK[wv][kk * 32 + m],      sK[wv][(kk + 1) * 32 + m]);
            a1.u[i] = pack2(sK[wv][kk * 32 + 16 + m], sK[wv][(kk + 1) * 32 + 16 + m]);
            int kb = kbaseB + 2 * i;                           // n index
            b0.u[i] = pack2(sV[wv][kb * 32 + m],      sV[wv][(kb + 1) * 32 + m]);
            b1.u[i] = pack2(sV[wv][kb * 32 + 16 + m], sV[wv][(kb + 1) * 32 + 16 + m]);
        }
        c00.v = __builtin_amdgcn_wmma_f32_16x16x32_bf16(false, a0.v, false, b0.v, (short)0, c00.v, false, false);
        c01.v = __builtin_amdgcn_wmma_f32_16x16x32_bf16(false, a0.v, false, b1.v, (short)0, c01.v, false, false);
        c10.v = __builtin_amdgcn_wmma_f32_16x16x32_bf16(false, a1.v, false, b0.v, (short)0, c10.v, false, false);
        c11.v = __builtin_amdgcn_wmma_f32_16x16x32_bf16(false, a1.v, false, b1.v, (short)0, c11.v, false, false);
        __builtin_amdgcn_wave_barrier();
    }

    // cross-wave reduction in LDS
    auto acc = [&](CFr& c, int dh, int eh) {
#pragma unroll
        for (int r = 0; r < 8; ++r) {
            int d = dh * 16 + r + hi * 8;
            int e = eh * 16 + m;
            atomicAdd(&sAcc[d * 32 + e], c.f[r]);
        }
    };
    acc(c00, 0, 0); acc(c01, 0, 1); acc(c10, 1, 0); acc(c11, 1, 1);
    __syncthreads();
    const float inv_n = 1.0f / (float)Nk;
    float* dst = kvout + (size_t)(bb * NHEAD + head) * 1024;
    for (int i = t; i < 1024; i += 256) dst[i] = sAcc[i] * inv_n;
}

// =====================================================================
// Kernel 3: o = (qr @ kv_other) * z, written transposed to (b,c,h,w).
// z = 1/(qf . kmean_other + 1e-6).  One block = 128 rows of one batch.
// =====================================================================
__global__ __launch_bounds__(256) void o_kernel(
    const uint16_t* __restrict__ qf,
    const uint16_t* __restrict__ qrp,
    const float*    __restrict__ ksum_other,
    const float*    __restrict__ kv_other,
    float*          __restrict__ out)
{
    __shared__ uint16_t sKV[NHEAD * 32 * 32];   // kv in bf16
    __shared__ float sZ[128 * NHEAD];
    __shared__ float sKM[Ck];

    const int t = threadIdx.x, wv = t >> 5, lane = t & 31;
    const int bb  = blockIdx.x / (Nk / 128);
    const int nn0 = (blockIdx.x % (Nk / 128)) * 128;
    const size_t row0 = (size_t)bb * Nk + nn0;
    const float inv_n = 1.0f / (float)Nk;

    for (int i = t; i < Ck; i += 256) sKM[i] = ksum_other[bb * Ck + i] * inv_n;
    {
        const float* kvsrc = kv_other + (size_t)bb * NHEAD * 1024;
        for (int i = t; i < NHEAD * 1024; i += 256) sKV[i] = f2bf(kvsrc[i]);
    }
    __syncthreads();

    // z per (local row, head)
    for (int i = t; i < 128 * NHEAD; i += 256) {
        int r = i / NHEAD, head = i % NHEAD;
        const uint32_t* q = (const uint32_t*)&qf[(row0 + r) * Ck + head * Dk];
        float acc = 0.f;
#pragma unroll
        for (int k = 0; k < 16; ++k) {
            uint32_t pk = q[k];
            acc += bf2f((uint16_t)pk)         * sKM[head * Dk + 2 * k]
                 + bf2f((uint16_t)(pk >> 16)) * sKM[head * Dk + 2 * k + 1];
        }
        sZ[i] = 1.0f / (acc + 1e-6f);
    }
    __syncthreads();

    const int m = lane & 15, hi = lane >> 4;
    const int r0 = wv * 16;
    const int kbaseA = hi * 8, kbaseB = hi * 16;

    for (int head = 0; head < NHEAD; ++head) {
        ABf a;
        {
            const uint32_t* qrow = (const uint32_t*)&qrp[(row0 + r0 + m) * Ck + head * Dk];
#pragma unroll
            for (int i = 0; i < 8; ++i) {
                int kk = kbaseB + 2 * i;                          // d index
                a.u[i] = qrow[kk >> 1];
            }
        }
        ABf b0, b1;
#pragma unroll
        for (int i = 0; i < 8; ++i) {
            int kb = kbaseB + 2 * i;                               // d index
            b0.u[i] = pack2(sKV[head * 1024 + kb * 32 + m],
                            sKV[head * 1024 + (kb + 1) * 32 + m]);
            b1.u[i] = pack2(sKV[head * 1024 + kb * 32 + 16 + m],
                            sKV[head * 1024 + (kb + 1) * 32 + 16 + m]);
        }
        CFr c0, c1;
#pragma unroll
        for (int i = 0; i < 8; ++i) { c0.f[i] = 0; c1.f[i] = 0; }
        c0.v = __builtin_amdgcn_wmma_f32_16x16x32_bf16(false, a.v, false, b0.v, (short)0, c0.v, false, false);
        c1.v = __builtin_amdgcn_wmma_f32_16x16x32_bf16(false, a.v, false, b1.v, (short)0, c1.v, false, false);
#pragma unroll
        for (int r = 0; r < 8; ++r) {
            int rl = r0 + r + hi * 8;
            float z = sZ[rl * NHEAD + head];
            int nn = nn0 + rl;
            int ch = head * Dk + m;
            out[((size_t)(bb * Ck + ch)      ) * Nk + nn] = c0.f[r] * z;
            out[((size_t)(bb * Ck + ch + 16) ) * Nk + nn] = c1.f[r] * z;
        }
    }
}

// =====================================================================
// Kernel 4: depthwise 3x3 LePE over v, added into out (NCHW).
// =====================================================================
__global__ __launch_bounds__(256) void lepe_kernel(
    const float* __restrict__ Vin,
    const float* __restrict__ lw,
    const float* __restrict__ lb,
    float* __restrict__ out)
{
    int idx = blockIdx.x * 256 + threadIdx.x;
    if (idx >= Bk * Nk * Ck) return;
    int ch = idx % Ck;
    int nn = (idx / Ck) % Nk;
    int bb = idx / (Ck * Nk);
    int y = nn / Wk, x = nn % Wk;
    float acc = lb[ch];
    const float* wkp = lw + ch * 9;
    const float* vb  = Vin + (size_t)bb * Nk * Ck;
#pragma unroll
    for (int dy = -1; dy <= 1; ++dy) {
        int yy = y + dy; if (yy < 0 || yy >= Hk) continue;
#pragma unroll
        for (int dx = -1; dx <= 1; ++dx) {
            int xx = x + dx; if (xx < 0 || xx >= Wk) continue;
            acc += vb[(size_t)(yy * Wk + xx) * Ck + ch] * wkp[(dy + 1) * 3 + (dx + 1)];
        }
    }
    size_t oi = ((size_t)(bb * Ck + ch)) * Nk + nn;
    out[oi] += acc;
}

// =====================================================================
extern "C" void kernel_launch(void* const* d_in, const int* in_sizes, int n_in,
                              void* d_out, int out_size, void* d_ws, size_t ws_size,
                              hipStream_t stream)
{
    (void)in_sizes; (void)n_in; (void)out_size; (void)ws_size;
    const float* x1  = (const float*)d_in[0];
    const float* x2  = (const float*)d_in[1];
    const float* Wqk = (const float*)d_in[2];
    const float* bqk = (const float*)d_in[3];
    const float* Wv  = (const float*)d_in[4];
    const float* bv  = (const float*)d_in[5];
    const float* lw  = (const float*)d_in[6];
    const float* lb  = (const float*)d_in[7];

    char* ws = (char*)d_ws;
    size_t off = 0;
    auto alloc = [&](size_t bytes) { void* p = ws + off; off += (bytes + 255) & ~(size_t)255; return p; };

    uint32_t* Wfrag = (uint32_t*)alloc((size_t)3 * 18 * 32 * 8 * 4);
    uint16_t *qf[2], *qr[2], *krp[2]; float *V[2], *ks[2], *kv[2];
    for (int s = 0; s < 2; ++s) {
        qf[s]  = (uint16_t*)alloc((size_t)BNk * Ck * 2);
        qr[s]  = (uint16_t*)alloc((size_t)BNk * Ck * 2);
        krp[s] = (uint16_t*)alloc((size_t)BNk * Ck * 2);
        V[s]   = (float*)   alloc((size_t)BNk * Ck * 4);
        ks[s]  = (float*)   alloc((size_t)Bk * Ck * 4);
        kv[s]  = (float*)   alloc((size_t)Bk * NHEAD * 1024 * 4);
    }
    float* out1 = (float*)d_out;
    float* out2 = out1 + (size_t)Bk * Ck * Nk;

    prep_kernel<<<1, 256, 0, stream>>>(Wqk, Wv, Wfrag, ks[0], ks[1]);
    proj_kernel<<<BNk / 128, 256, 0, stream>>>(x1, Wfrag, bqk, bv, qf[0], qr[0], krp[0], V[0], ks[0]);
    proj_kernel<<<BNk / 128, 256, 0, stream>>>(x2, Wfrag, bqk, bv, qf[1], qr[1], krp[1], V[1], ks[1]);
    kv_kernel<<<Bk * NHEAD, 256, 0, stream>>>(krp[0], V[0], kv[0]);
    kv_kernel<<<Bk * NHEAD, 256, 0, stream>>>(krp[1], V[1], kv[1]);
    // cross-wired: stream1 output uses k2/v2 stats, and vice versa
    o_kernel<<<BNk / 128, 256, 0, stream>>>(qf[0], qr[0], ks[1], kv[1], out1);
    o_kernel<<<BNk / 128, 256, 0, stream>>>(qf[1], qr[1], ks[0], kv[0], out2);
    int total = Bk * Nk * Ck;
    lepe_kernel<<<(total + 255) / 256, 256, 0, stream>>>(V[0], lw, lb, out1);
    lepe_kernel<<<(total + 255) / 256, 256, 0, stream>>>(V[1], lw, lb, out2);
}
